// Model_23244363006053
// MI455X (gfx1250) — matrix operand order repaired and checked by reading, not compiled, not run on hardware
//
#include <hip/hip_runtime.h>
#include <math.h>

typedef unsigned short u16;
typedef __attribute__((ext_vector_type(8)))  unsigned short us8;
typedef __attribute__((ext_vector_type(16))) unsigned short us16;
typedef __attribute__((ext_vector_type(16))) __bf16 v16bf;
typedef __attribute__((ext_vector_type(8)))  float v8f;

#define TEMP_  0.0005f
#define B_     32
#define T_     1500
#define S_     256
#define VOCAB_ 100
#define TCH_   512
#define HALO_  2            // max conv SAME-pad (k=5 -> 2); rows per batch = L+4

// ---------- bf16 helpers ----------
__device__ __forceinline__ u16 f2bf(float f) {
  unsigned int u = __float_as_uint(f);
  u += 0x7FFFu + ((u >> 16) & 1u);          // round-to-nearest-even
  return (u16)(u >> 16);
}
__device__ __forceinline__ float bf2f(u16 h) {
  return __uint_as_float(((unsigned int)h) << 16);
}
__device__ __forceinline__ v16bf pack_frag(us8 lo, us8 hi) {
  union { us16 u; v16bf b; } f;
#pragma unroll
  for (int i = 0; i < 8; ++i) { f.u[i] = lo[i]; f.u[i + 8] = hi[i]; }
  return f.b;
}

// ---------- prep kernels ----------
__global__ void __launch_bounds__(256)
zero_k(u16* __restrict__ p, int n) {
  int i = blockIdx.x * 256 + threadIdx.x;
  if (i < n) p[i] = 0;
}

// zero the 4 halo rows per batch of a [B][L+4][C] activation buffer
__global__ void __launch_bounds__(256)
zero_halo_k(u16* __restrict__ p, int L, int C) {
  int idx = blockIdx.x * 256 + threadIdx.x;
  int total = B_ * 4 * C;
  if (idx >= total) return;
  int c  = idx % C;
  int r4 = (idx / C) & 3;
  int b  = idx / (4 * C);
  int row = (r4 < 2) ? r4 : (L + 2 + (r4 - 2));   // rows 0,1 and L+2,L+3
  p[(size_t)(b * (L + 4) + row) * C + c] = 0;
}

// W f32 [Cout][Cin][K]  ->  Wt bf16 [dk][Cout][CinPad] (zero-padded Cin)
__global__ void __launch_bounds__(256)
wt_transform_k(const float* __restrict__ W, u16* __restrict__ Wt,
               int Cout, int Cin, int CinPad, int Ksize) {
  int idx = blockIdx.x * 256 + threadIdx.x;
  int total = Ksize * Cout * CinPad;
  if (idx >= total) return;
  int ci = idx % CinPad;
  int rem = idx / CinPad;
  int co = rem % Cout;
  int dk = rem / Cout;
  float v = (ci < Cin) ? W[((size_t)co * Cin + ci) * Ksize + dk] : 0.0f;
  Wt[idx] = f2bf(v);
}

// phonemes [B,1,S] int32 + embedding [VOCAB,512] f32 -> ph bf16 [B][S+4][512]
__global__ void __launch_bounds__(256)
embed_k(const int* __restrict__ ph, const float* __restrict__ emb,
        u16* __restrict__ out) {
  int idx = blockIdx.x * 256 + threadIdx.x;
  if (idx >= B_ * S_ * TCH_) return;
  int c  = idx & (TCH_ - 1);
  int bs = idx >> 9;
  int s  = bs & (S_ - 1);
  int b  = bs >> 8;
  int id = ph[bs];
  out[(size_t)(b * (S_ + 4) + HALO_ + s) * TCH_ + c] = f2bf(emb[(size_t)id * TCH_ + c]);
}

// mels f32 [B][80][T] -> bf16 [B][T+4][96] (channels padded with zeros)
__global__ void __launch_bounds__(256)
melsT_k(const float* __restrict__ mels, u16* __restrict__ out) {
  int idx = blockIdx.x * 256 + threadIdx.x;
  if (idx >= B_ * T_ * 96) return;
  int c  = idx % 96;
  int bt = idx / 96;
  int t  = bt % T_;
  int b  = bt / T_;
  float v = (c < 80) ? mels[((size_t)b * 80 + c) * T_ + t] : 0.0f;
  out[(size_t)(b * (T_ + 4) + HALO_ + t) * 96 + c] = f2bf(v);
}

// ---------- conv1d(SAME) as implicit GEMM via WMMA ----------
// Each wave: 16 positions x 64 output channels (4 accumulators sharing one A
// fragment -> 4 WMMA per A load). Grid: (posTiles, coutBlocks(256), batch).
// All boundary handling is a branchless clamp into zeroed halo rows, so the
// inner loop is an unconditional load/WMMA stream with EXEC all-1s.
// X  : bf16 [B][L+4][CINPAD]    (position-major, channel-contiguous, halo'd)
// Wt : bf16 [dk][Cout][CINPAD]
// Y  : bf16 [B][L+4][CoutStride]
template <int CINPAD, int KS>
__global__ void __launch_bounds__(128)
conv_wmma_k(const u16* __restrict__ X, const u16* __restrict__ Wt,
            const float* __restrict__ bias, u16* __restrict__ Y,
            int L, int Cout, int CoutStride, int doRelu) {
  const int lane = threadIdx.x & 31;
  const int wave = threadIdx.x >> 5;
  const int hi   = lane >> 4;
  const int col  = lane & 15;
  const int b    = blockIdx.z;
  const int s0   = blockIdx.x << 4;
  const int n0   = (blockIdx.y * 4 + wave) << 6;   // 64 columns per wave
  if (n0 >= Cout) return;                           // wave-uniform exit

  v8f acc[4];
#pragma unroll
  for (int j = 0; j < 4; ++j)
#pragma unroll
    for (int i = 0; i < 8; ++i) acc[j][i] = 0.0f;

  constexpr int pad = KS >> 1;
#pragma unroll
  for (int dk = 0; dk < KS; ++dk) {
    const int s_eff = min(s0 + col + dk - pad, L + 1);    // halo-clamped row
    const u16* xrow  = X + (size_t)(b * (L + 4) + HALO_ + s_eff) * CINPAD;
    const u16* wbase = Wt + (size_t)(dk * Cout + n0 + col) * CINPAD;
    if (dk + 1 < KS)
      __builtin_prefetch(xrow + CINPAD, 0, 1);            // next dk's row
#pragma unroll 2
    for (int c0 = 0; c0 < CINPAD; c0 += 32) {
      // A 16x32 bf16: lo-lanes K {0..7,16..23}, hi-lanes K {8..15,24..31}
      us8 alo = *(const us8*)(xrow + c0 + hi * 8);
      us8 ahi = *(const us8*)(xrow + c0 + 16 + hi * 8);
      v16bf a = pack_frag(alo, ahi);
#pragma unroll
      for (int j = 0; j < 4; ++j) {
        if (n0 + j * 16 < Cout) {                         // wave-uniform
          const u16* wrow = wbase + (size_t)(j * 16) * CINPAD + c0 + hi * 8;
          us8 blo = *(const us8*)(wrow);
          us8 bhi = *(const us8*)(wrow + 16);
          v16bf bm = pack_frag(blo, bhi);
          acc[j] = __builtin_amdgcn_wmma_f32_16x16x32_bf16(
              false, a, false, bm, (short)0, acc[j], false, false);
        }
      }
    }
  }

#pragma unroll
  for (int j = 0; j < 4; ++j) {
    const int nb = n0 + j * 16;
    if (nb < Cout) {
      const float bv = bias[nb + col];
#pragma unroll
      for (int r = 0; r < 8; ++r) {
        const int m  = r + hi * 8;                        // D: row = vgpr + 8*hi
        const int so = s0 + m;
        if (so < L) {
          float v = acc[j][r] + bv;
          if (doRelu) v = fmaxf(v, 0.0f);
          Y[(size_t)(b * (L + 4) + HALO_ + so) * CoutStride + nb + col] = f2bf(v);
        }
      }
    }
  }
}

// ---------- fused attention: qk WMMA + log_softmax + prior + mask ----------
// Q: bf16 [B][T+4][96], K: bf16 [B][S+4][96] (halo'd, channels 80..95 zero)
__global__ void __launch_bounds__(512)
attn_k(const u16* __restrict__ Q, const u16* __restrict__ K,
       const float* __restrict__ prior, const unsigned char* __restrict__ mask,
       float* __restrict__ out) {
  constexpr int CP = 96;
  __shared__ float q2s[16];
  __shared__ float k2s[S_];
  __shared__ float tile[16][S_];

  const int b    = blockIdx.y;
  const int t0   = blockIdx.x << 4;
  const int tid  = threadIdx.x;
  const int lane = tid & 31;
  const int wave = tid >> 5;

  // ||k||^2 per s and ||q||^2 per row into LDS
  if (tid < S_) {
    const u16* kr = K + (size_t)(b * (S_ + 4) + HALO_ + tid) * CP;
    float s = 0.0f;
    for (int c = 0; c < 80; ++c) { float v = bf2f(kr[c]); s += v * v; }
    k2s[tid] = s;
  } else if (tid < S_ + 16) {
    int m = tid - S_;
    int t = t0 + m;
    float s = 0.0f;
    if (t < T_) {
      const u16* qr = Q + (size_t)(b * (T_ + 4) + HALO_ + t) * CP;
      for (int c = 0; c < 80; ++c) { float v = bf2f(qr[c]); s += v * v; }
    }
    q2s[m] = s;
  }
  __syncthreads();

  // each of the 16 waves owns one 16x16 s-tile of the [16 x 256] row-block
  const int hi  = lane >> 4;
  const int col = lane & 15;
  const int sn0 = wave << 4;

  // branchless halo clamp for grid-tail t rows (clamped rows never stored)
  const int t_eff = min(t0 + col, T_ + 1);
  const u16* qr = Q + (size_t)(b * (T_ + 4) + HALO_ + t_eff) * CP;

  v8f acc;
#pragma unroll
  for (int i = 0; i < 8; ++i) acc[i] = 0.0f;

#pragma unroll
  for (int c0 = 0; c0 < CP; c0 += 32) {
    us8 alo = *(const us8*)(qr + c0 + hi * 8);
    us8 ahi = *(const us8*)(qr + c0 + 16 + hi * 8);
    v16bf a = pack_frag(alo, ahi);
    const u16* kr = K + (size_t)(b * (S_ + 4) + HALO_ + sn0 + col) * CP + c0 + hi * 8;
    us8 blo = *(const us8*)kr;
    us8 bhi = *(const us8*)(kr + 16);
    v16bf bm = pack_frag(blo, bhi);
    acc = __builtin_amdgcn_wmma_f32_16x16x32_bf16(false, a, false, bm,
                                                  (short)0, acc, false, false);
  }

#pragma unroll
  for (int r = 0; r < 8; ++r) {
    const int m = r + hi * 8;
    float logit = -TEMP_ * (q2s[m] + k2s[sn0 + col] - 2.0f * acc[r]);
    tile[m][sn0 + col] = logit;
  }
  __syncthreads();

  // wave w reduces row w (256 elems, 8 per lane) -> log_softmax + prior + mask
  {
    const int m = wave;
    const int t = t0 + m;
    float vals[8];
    float mx = -INFINITY;
#pragma unroll
    for (int i = 0; i < 8; ++i) {
      vals[i] = tile[m][lane + i * 32];
      mx = fmaxf(mx, vals[i]);
    }
    for (int off = 16; off >= 1; off >>= 1)
      mx = fmaxf(mx, __shfl_xor(mx, off, 32));
    float sum = 0.0f;
#pragma unroll
    for (int i = 0; i < 8; ++i) sum += __expf(vals[i] - mx);
    for (int off = 16; off >= 1; off >>= 1)
      sum += __shfl_xor(sum, off, 32);
    const float lse = mx + __logf(sum);
    if (t < T_) {
      const size_t base = (size_t)(b * T_ + t) * S_;
#pragma unroll
      for (int i = 0; i < 8; ++i) {
        int sidx = lane + i * 32;
        float x = vals[i] - lse + __logf(prior[base + sidx] + 1e-8f);
        if (!mask[base + sidx]) x = -INFINITY;
        out[base + sidx] = x;
      }
    }
  }
}

// ---------- host launch ----------
extern "C" void kernel_launch(void* const* d_in, const int* in_sizes, int n_in,
                              void* d_out, int out_size, void* d_ws, size_t ws_size,
                              hipStream_t stream) {
  (void)in_sizes; (void)n_in; (void)out_size; (void)ws_size;

  const int*           phonemes = (const int*)d_in[0];
  const float*         mels     = (const float*)d_in[1];
  const unsigned char* mask     = (const unsigned char*)d_in[2];
  const float*         prior    = (const float*)d_in[3];
  const float*         emb      = (const float*)d_in[4];
  const float* kw[5] = {(const float*)d_in[5], (const float*)d_in[7],
                        (const float*)d_in[9], (const float*)d_in[11],
                        (const float*)d_in[13]};
  const float* kb[5] = {(const float*)d_in[6], (const float*)d_in[8],
                        (const float*)d_in[10], (const float*)d_in[12],
                        (const float*)d_in[14]};
  const float* qw[3] = {(const float*)d_in[15], (const float*)d_in[17],
                        (const float*)d_in[19]};
  const float* qb[3] = {(const float*)d_in[16], (const float*)d_in[18],
                        (const float*)d_in[20]};

  char* ws = (char*)d_ws;
  size_t off = 0;
  auto alloc = [&](size_t elems) -> u16* {
    u16* p = (u16*)(ws + off);
    off = (off + elems * sizeof(u16) + 255) & ~(size_t)255;
    return p;
  };

  const int SR = S_ + 4;   // halo'd rows per batch (key side)
  const int TR = T_ + 4;   // halo'd rows per batch (query side)

  // transformed weights (bf16, [dk][Cout][CinPad])
  u16* wt0 = alloc((size_t)5 * 512 * 512);
  u16* wt1 = alloc((size_t)5 * 512 * 512);
  u16* wt2 = alloc((size_t)5 * 512 * 512);
  u16* wt3 = alloc((size_t)3 * 1024 * 512);
  u16* wt4 = alloc((size_t)1 * 80 * 1024);
  u16* wq0 = alloc((size_t)3 * 160 * 96);
  u16* wq1 = alloc((size_t)1 * 80 * 160);
  u16* wq2 = alloc((size_t)1 * 80 * 96);
  // activations (bf16, position-major, halo'd)
  u16* ph    = alloc((size_t)B_ * SR * 512);
  u16* kA    = alloc((size_t)B_ * SR * 1024);
  u16* kB    = alloc((size_t)B_ * SR * 1024);
  u16* kfin  = alloc((size_t)B_ * SR * 96);
  u16* melsT = alloc((size_t)B_ * TR * 96);
  u16* qA    = alloc((size_t)B_ * TR * 160);
  u16* qB    = alloc((size_t)B_ * TR * 96);
  u16* qfin  = alloc((size_t)B_ * TR * 96);

  auto gridFor = [](int n) { return (n + 255) / 256; };

  // fully zero channel-padded buffers (covers halos + pad channels)
  { int n = B_ * SR * 96; zero_k<<<gridFor(n), 256, 0, stream>>>(kfin, n); }
  { int n = B_ * TR * 96; zero_k<<<gridFor(n), 256, 0, stream>>>(qB, n); }
  { int n = B_ * TR * 96; zero_k<<<gridFor(n), 256, 0, stream>>>(qfin, n); }
  // zero halo rows of every (buffer, stride) used as conv input
  auto halo = [&](u16* p, int L, int C) {
    int n = B_ * 4 * C;
    zero_halo_k<<<gridFor(n), 256, 0, stream>>>(p, L, C);
  };
  halo(ph, S_, 512);
  halo(kA, S_, 512);
  halo(kB, S_, 512);
  halo(kB, S_, 1024);
  halo(melsT, T_, 96);
  halo(qA, T_, 160);

  // weight transforms
  auto wt_launch = [&](const float* W, u16* Wt, int Cout, int Cin, int CinPad, int Ks) {
    int total = Ks * Cout * CinPad;
    wt_transform_k<<<gridFor(total), 256, 0, stream>>>(W, Wt, Cout, Cin, CinPad, Ks);
  };
  wt_launch(kw[0], wt0, 512, 512, 512, 5);
  wt_launch(kw[1], wt1, 512, 512, 512, 5);
  wt_launch(kw[2], wt2, 512, 512, 512, 5);
  wt_launch(kw[3], wt3, 1024, 512, 512, 3);
  wt_launch(kw[4], wt4, 80, 1024, 1024, 1);
  wt_launch(qw[0], wq0, 160, 80, 96, 3);
  wt_launch(qw[1], wq1, 80, 160, 160, 1);
  wt_launch(qw[2], wq2, 80, 80, 96, 1);

  // embedding gather + mel transpose
  { int n = B_ * S_ * 512; embed_k<<<gridFor(n), 256, 0, stream>>>(phonemes, emb, ph); }
  { int n = B_ * T_ * 96;  melsT_k<<<gridFor(n), 256, 0, stream>>>(mels, melsT); }

  // conv layers: grid = (posTiles, ceil(Cout/256), B)
  auto cgrid = [](int L, int Cout) {
    return dim3((L + 15) / 16, (Cout + 255) / 256, B_);
  };
  // key encoder (L = 256)
  conv_wmma_k<512, 5><<<cgrid(S_, 512), 128, 0, stream>>>(ph, wt0, kb[0], kA, S_, 512, 512, 1);
  conv_wmma_k<512, 5><<<cgrid(S_, 512), 128, 0, stream>>>(kA, wt1, kb[1], kB, S_, 512, 512, 1);
  conv_wmma_k<512, 5><<<cgrid(S_, 512), 128, 0, stream>>>(kB, wt2, kb[2], kA, S_, 512, 512, 1);
  conv_wmma_k<512, 3><<<cgrid(S_, 1024), 128, 0, stream>>>(kA, wt3, kb[3], kB, S_, 1024, 1024, 1);
  conv_wmma_k<1024, 1><<<cgrid(S_, 80), 128, 0, stream>>>(kB, wt4, kb[4], kfin, S_, 80, 96, 0);
  // query encoder (L = 1500)
  conv_wmma_k<96, 3><<<cgrid(T_, 160), 128, 0, stream>>>(melsT, wq0, qb[0], qA, T_, 160, 160, 1);
  conv_wmma_k<160, 1><<<cgrid(T_, 80), 128, 0, stream>>>(qA, wq1, qb[1], qB, T_, 80, 96, 1);
  conv_wmma_k<96, 1><<<cgrid(T_, 80), 128, 0, stream>>>(qB, wq2, qb[2], qfin, T_, 80, 96, 0);

  // fused attention + log_softmax + prior + mask
  dim3 ag((T_ + 15) / 16, B_);
  attn_k<<<ag, 512, 0, stream>>>(qfin, kfin, prior, mask, (float*)d_out);
}
